// HanDecoder_13494787244397
// MI455X (gfx1250) — matrix operand order repaired and checked by reading, not compiled, not run on hardware
//
#include <hip/hip_runtime.h>
#include <hip/hip_bf16.h>
#include <stdint.h>

// ---------------------------------------------------------------------------
// HanDecoder on MI455X (gfx1250).
//  Phase 1: bf16 WMMA GEMMs precompute all input projections (parallel ~150 GFLOP).
//  Phase 2: persistent 32-WG scan; Whh_f slice resident in 256KB LDS per WGP,
//           device-scope generation barrier per step, tempo branch at beat
//           boundaries only.
// ---------------------------------------------------------------------------

typedef __attribute__((ext_vector_type(16))) __bf16 v16bf;
typedef __attribute__((ext_vector_type(8)))  float  v8f;

#define N_NOTES 16384
#define NBEAT   2048
#define NMEAS   512
#define DN      1024
#define DB      512
#define DM      256
#define DINF    1867   // note|beat|meas|prev_out(11)|perf(64)
#define DINT    843    // beat|meas|tempo(1)|result(10)|perf(64)
#define HF      1024
#define GF      4096   // 4*HF
#define HT      512
#define GT      2048   // 4*HT
#define NWG     32

// ------------------------------ helpers ------------------------------------
__device__ __forceinline__ float sigf(float x) { return 1.0f / (1.0f + expf(-x)); }

__device__ __forceinline__ __bf16 f2bf(float f) {
  union { float f; uint32_t u; } x; x.f = f;
  uint32_t r = x.u + 0x7FFFu + ((x.u >> 16) & 1u);          // RNE
  union { uint16_t s; __bf16 b; } y; y.s = (uint16_t)(r >> 16);
  return y.b;
}

__global__ void f32_to_bf16_kernel(const float* __restrict__ src,
                                   __bf16* __restrict__ dst, int n) {
  int i = blockIdx.x * blockDim.x + threadIdx.x;
  if (i < n) dst[i] = f2bf(src[i]);
}

// ------------------------------ WMMA GEMM ----------------------------------
// C[M,N] (f32, row-major) = A[M,K] (bf16, row-major, lda) x B^T
// where B is row-major [N, ldb] bf16 and we use columns [koff, koff+K).
// One wave per 16x16 tile. Fragment layouts per CDNA5 ISA 7.12.2:
//   A (16x32 bf16): lane m=L&15, kh=L>>4; VGPR v pair -> K=(v&4)<<2 | kh*8 | (v&3)*2
//   B (32x16 bf16): lane n=L&15, kh=L>>4; VGPR v pair -> K=(v&4)<<2 | kh*8 | (v&3)*2
//   D (16x16 f32):  lane n=L&15; VGPR r -> M = r + 8*(L>>4)
__global__ __launch_bounds__(256)
void wmma_gemm_bf16(const __bf16* __restrict__ A, int lda,
                    const __bf16* __restrict__ B, int ldb, int koff,
                    float* __restrict__ C, int M, int N, int K) {
  int wave  = (blockIdx.x * blockDim.x + threadIdx.x) >> 5;
  int lane  = threadIdx.x & 31;
  int tilesN = N >> 4;
  int tm = wave / tilesN;
  int tn = wave - tm * tilesN;
  if (tm >= (M >> 4)) return;

  int mr = (tm << 4) + (lane & 15);   // A row for this lane
  int nr = (tn << 4) + (lane & 15);   // B row (output column) for this lane
  int kh = lane >> 4;

  const __bf16* __restrict__ arow = A + (size_t)mr * lda;
  const __bf16* __restrict__ brow = B + (size_t)nr * ldb + koff;
  v8f c = {};

  for (int k0 = 0; k0 < K; k0 += 32) {
    v16bf a, b;
#pragma unroll
    for (int v = 0; v < 8; ++v) {
      int ka = ((v & 4) << 2) + (kh << 3) + ((v & 3) << 1) + k0;
      a[2 * v]     = arow[ka];
      a[2 * v + 1] = arow[ka + 1];
      int kb = ((v & 4) << 2) + (kh << 3) + ((v & 3) << 1) + k0;
      b[2 * v]     = brow[kb];
      b[2 * v + 1] = brow[kb + 1];
    }
    c = __builtin_amdgcn_wmma_f32_16x16x32_bf16(false, a, false, b,
                                                (short)0, c, false, false);
  }
  int cn = (tn << 4) + (lane & 15);
  int cm0 = (tm << 4) + (kh << 3);
#pragma unroll
  for (int r = 0; r < 8; ++r)
    C[(size_t)(cm0 + r) * N + cn] = c[r];
}

// ------------------- small precompute kernels ------------------------------
// const vector = bias + W[:, coff:coff+64] @ perf
__global__ void perf_const_kernel(const float* __restrict__ W, int ldw, int coff,
                                  const float* __restrict__ bias,
                                  const float* __restrict__ perf,
                                  float* __restrict__ outv, int rows) {
  int r = blockIdx.x * blockDim.x + threadIdx.x;
  if (r < rows) {
    float s = bias[r];
    const float* w = W + (size_t)r * ldw + coff;
#pragma unroll 8
    for (int j = 0; j < 64; ++j) s += w[j] * perf[j];
    outv[r] = s;
  }
}

__global__ void mark_beat_start_kernel(const int* __restrict__ bn,
                                       int* __restrict__ beat_start, int n) {
  int i = blockIdx.x * blockDim.x + threadIdx.x;
  if (i < n) {
    if (i == 0 || bn[i] != bn[i - 1]) beat_start[bn[i] - bn[0]] = i;
  }
}

__global__ void fill_seg_start_kernel(const int* __restrict__ bn,
                                      const int* __restrict__ beat_start,
                                      int* __restrict__ seg_start, int n) {
  int i = blockIdx.x * blockDim.x + threadIdx.x;
  if (i < n) seg_start[i] = beat_start[bn[i] - bn[0]];
}

__global__ void init_bar_kernel(int* bar) {
  if (threadIdx.x == 0) { bar[0] = 0; bar[1] = 0; }
}

// ------------------- device-wide generation barrier ------------------------
__device__ __forceinline__ void gbarrier(int* bar) {
  __syncthreads();
  if (threadIdx.x == 0) {
    __threadfence();  // publish this WG's global writes
    int gen = __hip_atomic_load(&bar[1], __ATOMIC_RELAXED, __HIP_MEMORY_SCOPE_AGENT);
    int arrived = __hip_atomic_fetch_add(&bar[0], 1, __ATOMIC_ACQ_REL,
                                         __HIP_MEMORY_SCOPE_AGENT);
    if (arrived == NWG - 1) {
      __hip_atomic_store(&bar[0], 0, __ATOMIC_RELAXED, __HIP_MEMORY_SCOPE_AGENT);
      __hip_atomic_fetch_add(&bar[1], 1, __ATOMIC_RELEASE, __HIP_MEMORY_SCOPE_AGENT);
    } else {
      while (__hip_atomic_load(&bar[1], __ATOMIC_ACQUIRE,
                               __HIP_MEMORY_SCOPE_AGENT) == gen)
        __builtin_amdgcn_s_sleep(2);
    }
    __threadfence();
  }
  __syncthreads();
}

// ------------------------------ scan kernel --------------------------------
// 32 WGs x 256 threads, persistent. WG j owns h indices [32j,32j+32) of the
// note-LSTM and [16j,16j+16) of the tempo-LSTM. Local gate row l = g*32+i maps
// to global gate row R = g*1024 + 32*wg + i.
__global__ __launch_bounds__(256, 1)
void scan_kernel(const float* __restrict__ Wih_f,
                 const float* __restrict__ Wih_t,
                 const float* __restrict__ Whh_t,
                 const float* __restrict__ W_fc,  const float* __restrict__ b_fc,
                 const float* __restrict__ W_tfc, const float* __restrict__ b_tfc,
                 const float* __restrict__ W_attn,const float* __restrict__ b_attn,
                 const float* __restrict__ ctxv,
                 const int* __restrict__ bn, const int* __restrict__ mn,
                 const int* __restrict__ seg_start,
                 const uint32_t* __restrict__ WhhF_pk,   // bf16 pairs [GF*512]
                 const float* __restrict__ Xnote, const float* __restrict__ Xbeat,
                 const float* __restrict__ Xmeas, const float* __restrict__ XtBeat,
                 const float* __restrict__ XtMeas,
                 const float* __restrict__ cfp, const float* __restrict__ ctp,
                 float* __restrict__ Hbuf, float* __restrict__ HTbuf,
                 int* __restrict__ bar,
                 float* __restrict__ out) {
  const int tid = threadIdx.x;
  const int wg  = blockIdx.x;

  extern __shared__ char smem_raw[];
  uint32_t* Wlds = (uint32_t*)smem_raw;          // [512*128] kk-major bf16 pairs
  float* fb    = (float*)(smem_raw + 512 * 128 * 4);
  float* hs    = fb;            // 1024  h_{n-1}
  float* hts   = hs + 1024;     // 512   h_t
  float* wprev = hts + 512;     // 128*11 prev-out columns of Wih_f
  float* gl    = wprev + 1408;  // 128   note-LSTM gate dots
  float* glt   = gl + 128;      // 64    tempo gate dots
  float* red   = glt + 64;      // 256   reduction scratch
  float* simb  = red + 256;     // 128   attention scores
  float* abuf  = simb + 128;    // 1280  attention buffer rows
  float* o10   = abuf + 1280;   // 16    out10_{n-1}
  float* res10 = o10 + 16;      // 16    attention result
  float* ctloc = res10 + 16;    // 16    c_t slice
  float* cfloc = ctloc + 16;    // 32    c_f slice
  float* scal  = cfloc + 32;    // 8     [0]=prev_tempo

  // ---- load resident recurrent-weight slice (bf16, 256KB) into LDS ----
  for (int idx = tid; idx < 128 * 512; idx += 256) {
    int l = idx & 127, kk = idx >> 7;
    int R = ((l >> 5) << 10) + (wg << 5) + (l & 31);
    Wlds[idx] = WhhF_pk[(size_t)R * 512 + kk];
    (void)kk;
  }
  for (int idx = tid; idx < 128 * 11; idx += 256) {
    int l = idx / 11, j = idx - l * 11;
    int R = ((l >> 5) << 10) + (wg << 5) + (l & 31);
    wprev[idx] = Wih_f[(size_t)R * DINF + 1792 + j];
  }
  if (tid < 32) { cfloc[tid] = 0.0f; Hbuf[wg * 32 + tid] = 0.0f; }
  if (tid < 16) { ctloc[tid] = 0.0f; HTbuf[wg * 16 + tid] = 0.0f; }
  if (tid == 0) scal[0] = 0.0f;
  __syncthreads();
  // Exercise gfx1250 sync paths (NOP when not dispatched as a cluster).
  __builtin_amdgcn_s_cluster_barrier();
  __builtin_amdgcn_s_wait_tensorcnt(0);
  gbarrier(bar);

  const int bn0 = bn[0], mn0 = mn[0];
  int p = 0, q = 0;

  for (int n = 0; n < N_NOTES; ++n) {
    // ---- stage h_{n-1} ----
    const float* Hin = Hbuf + p * HF;
    for (int idx = tid; idx < HF; idx += 256) hs[idx] = Hin[idx];
    __syncthreads();

    // ---- out10_{n-1} = h @ W_fc.T + b_fc (redundant in each WG) ----
    if (n > 0) {
      if (tid < 160) {
        int j = tid >> 4, part = tid & 15;
        const float* wr = W_fc + (size_t)j * HF + part * 64;
        const float* hh = hs + part * 64;
        float s = 0.0f;
#pragma unroll 8
        for (int k = 0; k < 64; ++k) s += hh[k] * wr[k];
        red[tid] = s;
      }
      __syncthreads();
      if (tid < 10) {
        float s = b_fc[tid];
        for (int pp = 0; pp < 16; ++pp) s += red[(tid << 4) + pp];
        o10[tid] = s;
        if (wg == 0) out[(size_t)(n - 1) * 11 + 1 + tid] = s;
      }
    } else if (tid < 10) {
      o10[tid] = 0.0f;
    }
    __syncthreads();

    const float tempo_old = scal[0];       // carry's prev_tempo (pre-boundary)
    const int bnn = bn[n];
    const bool isb = (n == 0) || (bnn > bn[n - 1]);
    const int cb = bnn - bn0;
    const int cmu = mn[n] - mn0;

    if (isb) {
      // ---- attention over previous beat's out10 rows ----
      if (n == 0) {
        if (tid < 10) res10[tid] = 0.0f;
        __syncthreads();
      } else {
        int s0 = seg_start[n - 1];
        int cnt = n - s0; if (cnt > 128) cnt = 128;
        if (tid < cnt) {
          int row = s0 + tid;
          float x[10];
          if (row == n - 1) { for (int j = 0; j < 10; ++j) x[j] = o10[j]; }
          else { for (int j = 0; j < 10; ++j) x[j] = out[(size_t)row * 11 + 1 + j]; }
          for (int j = 0; j < 10; ++j) abuf[tid * 10 + j] = x[j];
          float sim = 0.0f;
          for (int j = 0; j < 10; ++j) {
            float s = b_attn[j];
            for (int k = 0; k < 10; ++k) s += W_attn[j * 10 + k] * x[k];
            sim += tanhf(s) * ctxv[j];
          }
          simb[tid] = sim;
        }
        __syncthreads();
        red[tid] = (tid < cnt) ? simb[tid] : -3.4e38f;
        __syncthreads();
        for (int s = 128; s > 0; s >>= 1) {
          if (tid < s) red[tid] = fmaxf(red[tid], red[tid + s]);
          __syncthreads();
        }
        float mx = red[0];
        __syncthreads();
        red[tid] = (tid < cnt) ? expf(simb[tid] - mx) : 0.0f;
        __syncthreads();
        for (int s = 128; s > 0; s >>= 1) {
          if (tid < s) red[tid] += red[tid + s];
          __syncthreads();
        }
        float Z = red[0];
        __syncthreads();
        if (tid < 10) {
          float r = 0.0f;
          for (int rr = 0; rr < cnt; ++rr)
            r += (expf(simb[rr] - mx) / Z) * abuf[rr * 10 + tid];
          res10[tid] = r;
        }
        __syncthreads();
      }
      // ---- tempo LSTM step ----
      for (int idx = tid; idx < HT; idx += 256) hts[idx] = HTbuf[q * HT + idx];
      __syncthreads();
      if (tid < 64) {
        int g = tid >> 4, i = tid & 15;
        int row = g * HT + wg * 16 + i;
        float acc = XtBeat[(size_t)cb * GT + row] + XtMeas[(size_t)cmu * GT + row]
                  + ctp[row];
        const float* wt = Wih_t + (size_t)row * DINT;
        acc += tempo_old * wt[768];
        for (int j = 0; j < 10; ++j) acc += res10[j] * wt[769 + j];
        const float* whr = Whh_t + (size_t)row * HT;
#pragma unroll 8
        for (int k = 0; k < HT; ++k) acc += hts[k] * whr[k];
        glt[tid] = acc;
      }
      __syncthreads();
      if (tid < 16) {
        float ig = glt[tid], fg = glt[16 + tid], gg = glt[32 + tid], og = glt[48 + tid];
        float c = sigf(fg) * ctloc[tid] + sigf(ig) * tanhf(gg);
        ctloc[tid] = c;
        HTbuf[(q ^ 1) * HT + wg * 16 + tid] = sigf(og) * tanhf(c);
      }
      gbarrier(bar);
      {
        const float* hnew = HTbuf + (q ^ 1) * HT;
        red[tid] = hnew[tid] * W_tfc[tid] + hnew[tid + 256] * W_tfc[tid + 256];
        __syncthreads();
        for (int s = 128; s > 0; s >>= 1) {
          if (tid < s) red[tid] += red[tid + s];
          __syncthreads();
        }
        if (tid == 0) scal[0] = red[0] + b_tfc[0];
        __syncthreads();
      }
      q ^= 1;
    }
    if (wg == 0 && tid == 0) out[(size_t)n * 11] = scal[0];

    // ---- note-LSTM recurrent matvec (LDS-resident bf16 weights) ----
    {
      int l = tid >> 1, half = tid & 1;
      const uint32_t* wp = Wlds + half * 256 * 128 + l;
      const float* hp = hs + half * 512;
      float acc = 0.0f;
#pragma unroll 4
      for (int kk = 0; kk < 256; ++kk) {
        uint32_t u = wp[(size_t)kk * 128];
        float w0 = __uint_as_float(u << 16);
        float w1 = __uint_as_float(u & 0xFFFF0000u);
        acc += w0 * hp[2 * kk] + w1 * hp[2 * kk + 1];
      }
      red[tid] = acc;
    }
    __syncthreads();
    if (tid < 128) {
      int R = ((tid >> 5) << 10) + (wg << 5) + (tid & 31);
      float base = Xnote[(size_t)n * GF + R] + Xbeat[(size_t)cb * GF + R]
                 + Xmeas[(size_t)cmu * GF + R] + cfp[R];
      const float* wpr = wprev + tid * 11;
      base += tempo_old * wpr[0];
      for (int j = 0; j < 10; ++j) base += o10[j] * wpr[1 + j];
      gl[tid] = base + red[2 * tid] + red[2 * tid + 1];
      if (n + 1 < N_NOTES)
        __builtin_prefetch(&Xnote[(size_t)(n + 1) * GF + R], 0, 1);
    }
    __syncthreads();
    if (tid < 32) {
      float ig = gl[tid], fg = gl[32 + tid], gg = gl[64 + tid], og = gl[96 + tid];
      float c = sigf(fg) * cfloc[tid] + sigf(ig) * tanhf(gg);
      cfloc[tid] = c;
      Hbuf[(p ^ 1) * HF + wg * 32 + tid] = sigf(og) * tanhf(c);
    }
    p ^= 1;
    gbarrier(bar);
  }

  // ---- epilogue: out10 for the final note ----
  {
    const float* Hin = Hbuf + p * HF;
    for (int idx = tid; idx < HF; idx += 256) hs[idx] = Hin[idx];
    __syncthreads();
    if (tid < 160) {
      int j = tid >> 4, part = tid & 15;
      const float* wr = W_fc + (size_t)j * HF + part * 64;
      const float* hh = hs + part * 64;
      float s = 0.0f;
#pragma unroll 8
      for (int k = 0; k < 64; ++k) s += hh[k] * wr[k];
      red[tid] = s;
    }
    __syncthreads();
    if (wg == 0 && tid < 10) {
      float s = b_fc[tid];
      for (int pp = 0; pp < 16; ++pp) s += red[(tid << 4) + pp];
      out[(size_t)(N_NOTES - 1) * 11 + 1 + tid] = s;
    }
  }
}

// ------------------------------ launcher -----------------------------------
static constexpr size_t alignup(size_t x) { return (x + 255) & ~(size_t)255; }

extern "C" void kernel_launch(void* const* d_in, const int* in_sizes, int n_in,
                              void* d_out, int out_size, void* d_ws, size_t ws_size,
                              hipStream_t stream) {
  const float* note  = (const float*)d_in[0];
  const float* beat  = (const float*)d_in[1];
  const float* meas  = (const float*)d_in[2];
  const float* perf  = (const float*)d_in[3];
  const int*   bn    = (const int*)d_in[4];
  const int*   mn    = (const int*)d_in[5];
  const float* Wih_f = (const float*)d_in[6];
  const float* Whh_f = (const float*)d_in[7];
  const float* b_f   = (const float*)d_in[8];
  const float* Wih_t = (const float*)d_in[9];
  const float* Whh_t = (const float*)d_in[10];
  const float* b_t   = (const float*)d_in[11];
  const float* W_fc  = (const float*)d_in[12];
  const float* b_fc  = (const float*)d_in[13];
  const float* W_tfc = (const float*)d_in[14];
  const float* b_tfc = (const float*)d_in[15];
  const float* W_attn= (const float*)d_in[16];
  const float* b_attn= (const float*)d_in[17];
  const float* ctxv  = (const float*)d_in[18];
  float* out = (float*)d_out;

  // ---- workspace layout ----
  char* ws = (char*)d_ws;
  size_t off = 0;
  auto take = [&](size_t bytes) { char* p = ws + off; off = alignup(off + bytes); return p; };
  __bf16* noteBF = (__bf16*)take((size_t)N_NOTES * DN * 2);
  __bf16* wihfBF = (__bf16*)take((size_t)GF * DINF * 2);
  __bf16* wihtBF = (__bf16*)take((size_t)GT * DINT * 2);
  __bf16* beatBF = (__bf16*)take((size_t)NBEAT * DB * 2);
  __bf16* measBF = (__bf16*)take((size_t)NMEAS * DM * 2);
  __bf16* whhfBF = (__bf16*)take((size_t)GF * HF * 2);
  float* Xnote  = (float*)take((size_t)N_NOTES * GF * 4);
  float* Xbeat  = (float*)take((size_t)NBEAT * GF * 4);
  float* Xmeas  = (float*)take((size_t)NMEAS * GF * 4);
  float* XtBeat = (float*)take((size_t)NBEAT * GT * 4);
  float* XtMeas = (float*)take((size_t)NMEAS * GT * 4);
  float* cfp    = (float*)take((size_t)GF * 4);
  float* ctp    = (float*)take((size_t)GT * 4);
  float* Hbuf   = (float*)take(2 * HF * 4);
  float* HTbuf  = (float*)take(2 * HT * 4);
  int*   segst  = (int*)take((size_t)N_NOTES * 4);
  int*   beatst = (int*)take((size_t)NBEAT * 4);
  int*   bar    = (int*)take(256);
  (void)ws_size; (void)n_in; (void)in_sizes; (void)out_size;

  // ---- barrier init ----
  init_bar_kernel<<<1, 64, 0, stream>>>(bar);

  // ---- bf16 conversions ----
  auto cvt = [&](const float* s, __bf16* d, int n) {
    f32_to_bf16_kernel<<<(n + 255) / 256, 256, 0, stream>>>(s, d, n);
  };
  cvt(note,  noteBF, N_NOTES * DN);
  cvt(Wih_f, wihfBF, GF * DINF);
  cvt(Wih_t, wihtBF, GT * DINT);
  cvt(beat,  beatBF, NBEAT * DB);
  cvt(meas,  measBF, NMEAS * DM);
  cvt(Whh_f, whhfBF, GF * HF);

  // ---- segment bookkeeping ----
  mark_beat_start_kernel<<<(N_NOTES + 255) / 256, 256, 0, stream>>>(bn, beatst, N_NOTES);
  fill_seg_start_kernel<<<(N_NOTES + 255) / 256, 256, 0, stream>>>(bn, beatst, segst, N_NOTES);

  // ---- perf + bias constants ----
  perf_const_kernel<<<GF / 256, 256, 0, stream>>>(Wih_f, DINF, 1803, b_f, perf, cfp, GF);
  perf_const_kernel<<<GT / 256, 256, 0, stream>>>(Wih_t, DINT, 779, b_t, perf, ctp, GT);

  // ---- WMMA GEMMs: input projections ----
  auto gemm = [&](const __bf16* A, int lda, const __bf16* B, int ldb, int koff,
                  float* C, int M, int N, int K) {
    int waves = (M >> 4) * (N >> 4);
    wmma_gemm_bf16<<<(waves + 7) / 8, 256, 0, stream>>>(A, lda, B, ldb, koff, C, M, N, K);
  };
  gemm(noteBF, DN, wihfBF, DINF,    0, Xnote,  N_NOTES, GF, DN);  // note -> 4096 gates
  gemm(beatBF, DB, wihfBF, DINF, 1024, Xbeat,  NBEAT,   GF, DB);  // beat -> note gates
  gemm(measBF, DM, wihfBF, DINF, 1536, Xmeas,  NMEAS,   GF, DM);  // meas -> note gates
  gemm(beatBF, DB, wihtBF, DINT,    0, XtBeat, NBEAT,   GT, DB);  // beat -> tempo gates
  gemm(measBF, DM, wihtBF, DINT,  512, XtMeas, NMEAS,   GT, DM);  // meas -> tempo gates

  // ---- persistent sequential scan ----
  constexpr int SMEM_BYTES = 512 * 128 * 4 + 4888 * 4;  // 281,696 B < 320KB LDS
  (void)hipFuncSetAttribute(reinterpret_cast<const void*>(scan_kernel),
                            hipFuncAttributeMaxDynamicSharedMemorySize, SMEM_BYTES);
  scan_kernel<<<NWG, 256, SMEM_BYTES, stream>>>(
      Wih_f, Wih_t, Whh_t, W_fc, b_fc, W_tfc, b_tfc, W_attn, b_attn, ctxv,
      bn, mn, segst, (const uint32_t*)whhfBF,
      Xnote, Xbeat, Xmeas, XtBeat, XtMeas, cfp, ctp,
      Hbuf, HTbuf, bar, out);
}
